// SWA_56049323213662
// MI455X (gfx1250) — matrix operand order repaired and checked by reading, not compiled, not run on hardware
//
#include <hip/hip_runtime.h>
#include <hip/hip_bf16.h>

typedef __attribute__((ext_vector_type(16))) _Float16 v16h;
typedef __attribute__((ext_vector_type(8)))  _Float16 v8h;
typedef __attribute__((ext_vector_type(8)))  float    v8f;
typedef __attribute__((ext_vector_type(4)))  float    v4f;
typedef __attribute__((ext_vector_type(4)))  int      v4i;

#define BATCH 2
#define HH    56
#define WW    56
#define CC    256
#define NH    8
#define DK    32
#define HWQ   3136      // 56*56
#define KV    3844      // 62*62
#define KVP   3872      // padded to 32*121
#define NSTEP 121
#define QT    196       // 3136/16
#define KT    242       // 3872/16
#define LPITCH 40       // LDS row pitch in f16 (80B: 16B-aligned, conflict-friendly)

#if defined(__has_builtin)
#if __has_builtin(__builtin_amdgcn_global_load_async_to_lds_b128) && \
    __has_builtin(__builtin_amdgcn_s_wait_asynccnt)
#define USE_ASYNC 1
#endif
#endif
#ifndef USE_ASYNC
#define USE_ASYNC 0
#endif

#if USE_ASYNC
typedef __attribute__((address_space(1))) v4i as1_v4i;   // prints as "int4 __device__ *"
typedef __attribute__((address_space(3))) v4i as3_v4i;   // prints as "int4 __shared__ *"
#endif

static __device__ __forceinline__ int refl(int p) {
    int i = p - 3;
    if (i < 0)  i = -i;
    if (i > 55) i = 110 - i;
    return i;
}

// ---------------------------------------------------------------------------
// Kernel 0: transpose + convert weights to f16: Wt[mat][out][in]
// ---------------------------------------------------------------------------
__global__ void swa_wtrans(const float* __restrict__ Wq,
                           const float* __restrict__ Wk,
                           const float* __restrict__ Wv,
                           _Float16* __restrict__ Wt) {
    int idx = blockIdx.x * blockDim.x + threadIdx.x;   // < 3*65536
    int mat = idx >> 16;
    int r   = idx & 65535;
    int o   = r >> 8;          // out channel
    int i   = r & 255;         // in channel
    const float* W = (mat == 0) ? Wq : (mat == 1 ? Wk : Wv);
    Wt[idx] = (_Float16)W[i * 256 + o];   // Wt[mat][o][i] = W[i][o]
}

// ---------------------------------------------------------------------------
// Kernel 1: QKV projection via WMMA. One wave = 16 rows x 64 out-channels.
//  Q  -> Qh[b][pixel][256]  (f16, row-major)
//  K  -> Kh[b][kpix ][256]  (f16, row-major, kpix in padded 3872 space)
//  V  -> Vt[b][ch   ][3872] (f16, key-major / transposed)
// ---------------------------------------------------------------------------
__global__ __launch_bounds__(32)
void swa_qkv(const float* __restrict__ x,
             const _Float16* __restrict__ Wt,
             const float* __restrict__ bq,
             const float* __restrict__ bk,
             const float* __restrict__ bv,
             _Float16* __restrict__ Qh,
             _Float16* __restrict__ Kh,
             _Float16* __restrict__ Vt) {
    const int lane = threadIdx.x;
    int unit = blockIdx.x;
    int mat, b, rt;
    if (unit < 2 * QT)            { mat = 0; b = unit / QT; rt = unit % QT; }
    else if (unit < 2*QT + 2*KT)  { int u = unit - 2*QT; mat = 1; b = u / KT; rt = u % KT; }
    else                          { int u = unit - 2*QT - 2*KT; mat = 2; b = u / KT; rt = u % KT; }

    const _Float16* Wm  = Wt + (size_t)mat * 65536;
    const float*    bia = (mat == 0) ? bq : (mat == 1 ? bk : bv);
    const int nb = blockIdx.y * 64;

    const int rr = rt * 16 + (lane & 15);
    bool valid = true;
    int  src   = rr;
    if (mat != 0) {
        valid = (rr < KV);
        if (valid) {
            int py = rr / 62, px = rr % 62;
            src = refl(py) * WW + refl(px);
        } else {
            src = 0;
        }
    }
    const float* xr = x + ((size_t)b * HWQ + src) * CC;
    const int hh  = (lane & 16) ? 8  : 0;
    const int bh_ = (lane & 16) ? 16 : 0;

    v8f acc[4] = {v8f{}, v8f{}, v8f{}, v8f{}};

    for (int kc = 0; kc < 8; ++kc) {
        const int k0 = kc * 32;
        v16h a;
        if (valid) {
            v4f f0 = *(const v4f*)(xr + k0 + hh);
            v4f f1 = *(const v4f*)(xr + k0 + hh + 4);
            v4f f2 = *(const v4f*)(xr + k0 + 16 + hh);
            v4f f3 = *(const v4f*)(xr + k0 + 16 + hh + 4);
            #pragma unroll
            for (int j = 0; j < 4; ++j) {
                a[j]      = (_Float16)f0[j];
                a[4 + j]  = (_Float16)f1[j];
                a[8 + j]  = (_Float16)f2[j];
                a[12 + j] = (_Float16)f3[j];
            }
        } else {
            #pragma unroll
            for (int j = 0; j < 16; ++j) a[j] = (_Float16)0.f;
        }
        #pragma unroll
        for (int t = 0; t < 4; ++t) {
            const int col = nb + t * 16 + (lane & 15);
            const _Float16* wr = Wm + (size_t)col * 256 + k0 + hh;
            v8h lo = *(const v8h*)wr;
            v8h hi = *(const v8h*)(wr + 16);
            v16h bf;
            #pragma unroll
            for (int j = 0; j < 8; ++j) { bf[j] = lo[j]; bf[8 + j] = hi[j]; }
            acc[t] = __builtin_amdgcn_wmma_f32_16x16x32_f16(
                false, a, false, bf, (short)0, acc[t], false, false);
        }
    }

    const int rof = (lane & 16) ? 8 : 0;
    #pragma unroll
    for (int t = 0; t < 4; ++t) {
        const int outch = nb + t * 16 + (lane & 15);
        const float bval = bia[outch];
        #pragma unroll
        for (int i = 0; i < 8; ++i) {
            const int row = rt * 16 + i + rof;
            float val = acc[t][i] + bval;
            if (mat == 0) {
                Qh[((size_t)b * HWQ + row) * CC + outch] = (_Float16)val;
            } else {
                const bool rv = (row < KV);
                _Float16 h16 = (_Float16)(rv ? val : 0.f);
                if (mat == 1) Kh[((size_t)b * KVP + row) * CC + outch] = h16;
                else          Vt[((size_t)b * CC + outch) * KVP + row] = h16;
            }
        }
    }
}

// ---------------------------------------------------------------------------
// Kernel 2: fused flash attention.
// Workgroup = 4 waves sharing one (b, head); each wave owns a 16-query tile.
// K/V 32-key step tiles staged in LDS (async copy, double buffered).
// ---------------------------------------------------------------------------
__global__ __launch_bounds__(128)
void swa_attn(const _Float16* __restrict__ Qh,
              const _Float16* __restrict__ Kh,
              const _Float16* __restrict__ Vt,
              const float* __restrict__ x,
              const float* __restrict__ gptr,
              float* __restrict__ out) {
    __shared__ __align__(16) _Float16 kbuf[2][32][LPITCH];
    __shared__ __align__(16) _Float16 vbuf[2][32][LPITCH];

    const int tid  = threadIdx.x;
    const int lane = tid & 31;
    const int wid  = tid >> 5;

    int id = blockIdx.x;
    const int qt0 = (id % 49) * 4;  id /= 49;
    const int h   = id % NH;
    const int b   = id / NH;
    const int qt  = qt0 + wid;

    const int hh = (lane & 16) ? 8 : 0;

    // global bases for this (b, head)
    const _Float16* Kb = Kh + ((size_t)b * KVP) * CC + h * DK;
    const _Float16* Vb = Vt + ((size_t)b * CC + h * DK) * KVP;

    // staging assignment: 128 threads x (one 16B K chunk + one 16B V chunk)
    const int srow = tid >> 2;          // 0..31
    const int schk = (tid & 3) * 8;     // f16 offset of 16B chunk
    const _Float16* gK = Kb + (size_t)srow * CC  + schk;   // + step*32 rows
    const _Float16* gV = Vb + (size_t)srow * KVP + schk;   // + step*32 cols

    // --- Q as B-matrix fragment (32 dk x 16 queries), loaded once ---
    const int qcol = qt * 16 + (lane & 15);
    const _Float16* qr = Qh + ((size_t)b * HWQ + qcol) * CC + h * DK + hh;
    v16h qb;
    {
        v8h lo = *(const v8h*)qr;
        v8h hi = *(const v8h*)(qr + 16);
        #pragma unroll
        for (int j = 0; j < 8; ++j) { qb[j] = lo[j]; qb[8 + j] = hi[j]; }
    }

    v8f o0 = v8f{}, o1 = v8f{};
    float m = -1e30f, lsum = 0.f;

#if USE_ASYNC
    // prologue: issue step 0 into buffer 0
    __builtin_amdgcn_global_load_async_to_lds_b128(
        (as1_v4i*)gK, (as3_v4i*)&kbuf[0][srow][schk], 0, 0);
    __builtin_amdgcn_global_load_async_to_lds_b128(
        (as1_v4i*)gV, (as3_v4i*)&vbuf[0][srow][schk], 0, 0);
#endif

    for (int s = 0; s < NSTEP; ++s) {
        const int kbase = s * 32;
#if USE_ASYNC
        const int cur = s & 1;
        __builtin_amdgcn_s_wait_asynccnt(0);
        __syncthreads();                       // all waves' stages complete
        if (s + 1 < NSTEP) {                   // prefetch next step
            __builtin_amdgcn_global_load_async_to_lds_b128(
                (as1_v4i*)(gK + (size_t)(kbase + 32) * CC),
                (as3_v4i*)&kbuf[cur ^ 1][srow][schk], 0, 0);
            __builtin_amdgcn_global_load_async_to_lds_b128(
                (as1_v4i*)(gV + (kbase + 32)),
                (as3_v4i*)&vbuf[cur ^ 1][srow][schk], 0, 0);
        }
#else
        const int cur = 0;
        v8h tk = *(const v8h*)(gK + (size_t)kbase * CC);
        v8h tv = *(const v8h*)(gV + kbase);
        __syncthreads();                       // previous step's reads done
        *(v8h*)&kbuf[0][srow][schk] = tk;
        *(v8h*)&vbuf[0][srow][schk] = tv;
        __syncthreads();
#endif

        // --- K tiles as A-matrix fragments (16 keys x 32 dk) from LDS ---
        v16h ka0, ka1;
        {
            const _Float16* k0p = &kbuf[cur][lane & 15][0];
            const _Float16* k1p = &kbuf[cur][16 + (lane & 15)][0];
            v8h a0 = *(const v8h*)(k0p + hh), a1 = *(const v8h*)(k0p + 16 + hh);
            v8h c0 = *(const v8h*)(k1p + hh), c1 = *(const v8h*)(k1p + 16 + hh);
            #pragma unroll
            for (int j = 0; j < 8; ++j) {
                ka0[j] = a0[j]; ka0[8 + j] = a1[j];
                ka1[j] = c0[j]; ka1[8 + j] = c1[j];
            }
        }
        // S^T tiles: M=key, N=query
        v8f st0 = __builtin_amdgcn_wmma_f32_16x16x32_f16(false, ka0, false, qb, (short)0, v8f{}, false, false);
        v8f st1 = __builtin_amdgcn_wmma_f32_16x16x32_f16(false, ka1, false, qb, (short)0, v8f{}, false, false);

        if (s == NSTEP - 1) {   // mask keys >= 3844 (padded KV tail)
            const int kb0 = kbase + hh;
            #pragma unroll
            for (int i = 0; i < 8; ++i) {
                if (kb0 + i >= KV) st0[i] = -1e30f;
                st1[i] = -1e30f;                    // kbase+16+.. >= 3856 always
            }
        }

        // --- online softmax (query column = lane&15; halves pair via xor16) ---
        float mloc = -1e30f;
        #pragma unroll
        for (int i = 0; i < 8; ++i) { mloc = fmaxf(mloc, st0[i]); mloc = fmaxf(mloc, st1[i]); }
        mloc = fmaxf(mloc, __shfl_xor(mloc, 16, 32));
        const float mnew  = fmaxf(m, mloc);
        const float alpha = __expf(m - mnew);
        m = mnew;

        union { v8h h; unsigned u[4]; } a0u, a1u, e0u, e1u;
        float sl = 0.f;
        #pragma unroll
        for (int i = 0; i < 8; ++i) {
            float p0 = __expf(st0[i] - mnew);
            float p1 = __expf(st1[i] - mnew);
            sl += p0 + p1;
            a0u.h[i] = (_Float16)p0;
            a1u.h[i] = (_Float16)p1;
        }
        sl += __shfl_xor(sl, 16, 32);
        lsum = lsum * alpha + sl;
        #pragma unroll
        for (int i = 0; i < 8; ++i) { o0[i] *= alpha; o1[i] *= alpha; }

        // exchange packed f16 probabilities with partner lane (8 dword shfls)
        #pragma unroll
        for (int j = 0; j < 4; ++j) {
            e0u.u[j] = __shfl_xor(a0u.u[j], 16, 32);
            e1u.u[j] = __shfl_xor(a1u.u[j], 16, 32);
        }
        const bool up = (lane & 16) != 0;
        v16h pb;   // P^T as B-matrix (32 keys x 16 queries)
        #pragma unroll
        for (int i = 0; i < 8; ++i) {
            pb[i]     = up ? e1u.h[i] : a0u.h[i];   // K 0-7  / 16-23
            pb[8 + i] = up ? a1u.h[i] : e0u.h[i];   // K 8-15 / 24-31
        }

        // --- V tiles as A-matrix fragments (16 dk x 32 keys) from LDS ---
        v16h va0, va1;
        {
            const _Float16* v0p = &vbuf[cur][lane & 15][0];
            const _Float16* v1p = &vbuf[cur][16 + (lane & 15)][0];
            v8h a0 = *(const v8h*)(v0p + 2 * hh), a1 = *(const v8h*)(v0p + 2 * hh + 8);
            v8h c0 = *(const v8h*)(v1p + 2 * hh), c1 = *(const v8h*)(v1p + 2 * hh + 8);
            #pragma unroll
            for (int j = 0; j < 8; ++j) {
                va0[j] = a0[j]; va0[8 + j] = a1[j];
                va1[j] = c0[j]; va1[8 + j] = c1[j];
            }
        }
        // O^T tiles: M=dk, N=query; alpha rescale already applied
        o0 = __builtin_amdgcn_wmma_f32_16x16x32_f16(false, va0, false, pb, (short)0, o0, false, false);
        o1 = __builtin_amdgcn_wmma_f32_16x16x32_f16(false, va1, false, pb, (short)0, o1, false, false);
    }

    // --- epilogue: out = gamma * (O / lsum) + x ---
    const float inv = 1.0f / lsum;
    const float g   = gptr[0];
    const size_t base = ((size_t)b * HWQ + qcol) * CC + h * DK;
    const float* xr = x + base;
    float* orow = out + base;
    const int dof = (lane & 16) ? 8 : 0;
    #pragma unroll
    for (int i = 0; i < 8; ++i) {
        int d0 = i + dof;          // o0: dk rows 0-15
        int d1 = 16 + i + dof;     // o1: dk rows 16-31
        orow[d0] = g * (o0[i] * inv) + xr[d0];
        orow[d1] = g * (o1[i] * inv) + xr[d1];
    }
}

// ---------------------------------------------------------------------------
extern "C" void kernel_launch(void* const* d_in, const int* in_sizes, int n_in,
                              void* d_out, int out_size, void* d_ws, size_t ws_size,
                              hipStream_t stream) {
    const float* x  = (const float*)d_in[0];
    const float* Wq = (const float*)d_in[1];
    const float* bq = (const float*)d_in[2];
    const float* Wk = (const float*)d_in[3];
    const float* bk = (const float*)d_in[4];
    const float* Wv = (const float*)d_in[5];
    const float* bv = (const float*)d_in[6];
    const float* gm = (const float*)d_in[7];
    float* out = (float*)d_out;

    // workspace layout (f16 elements)
    _Float16* wsh = (_Float16*)d_ws;
    _Float16* Wt = wsh;                                  // 3*256*256   = 196608
    _Float16* Qh = Wt + 3 * 256 * 256;                   // 2*3136*256  = 1605632
    _Float16* Kh = Qh + (size_t)BATCH * HWQ * CC;        // 2*3872*256  = 1982464
    _Float16* Vt = Kh + (size_t)BATCH * KVP * CC;        // 2*256*3872  = 1982464

    // 0) weight transpose+convert: 196608 elems
    swa_wtrans<<<768, 256, 0, stream>>>(Wq, Wk, Wv, Wt);

    // 1) QKV projection: units = 2*196 (Q) + 2*242 (K) + 2*242 (V) = 1360
    dim3 g1(2 * QT + 4 * KT, 4);
    swa_qkv<<<g1, 32, 0, stream>>>(x, Wt, bq, bk, bv, Qh, Kh, Vt);

    // 2) fused attention: (2*8*49) blocks x 4 waves (one (b,h) per block)
    swa_attn<<<BATCH * NH * 49, 128, 0, stream>>>(Qh, Kh, Vt, x, gm, out);
}
